// FlaxFalcon3Attention_23716809409287
// MI455X (gfx1250) — matrix operand order repaired and checked by reading, not compiled, not run on hardware
//
#include <hip/hip_runtime.h>

typedef __attribute__((ext_vector_type(16))) __bf16 v16bf;
typedef __attribute__((ext_vector_type(8)))  __bf16 v8bf;
typedef __attribute__((ext_vector_type(8)))  float  v8f;

#define S_LEN 4096
#define E_DIM 3072
#define HQ    12
#define HKV   4
#define DHEAD 256
#define NQK   4096          // q cols (3072) + k cols (1024) boundary
#define NALL  5120          // 3072 + 1024 + 1024
#define THETA_F 1000042.0f

// ---------- bf16 helpers (bit-exact ushort storage, RNE convert) ----------
static __device__ inline unsigned short f2bf(float f) {
  unsigned int u = __builtin_bit_cast(unsigned int, f);
  if ((u & 0x7f800000u) == 0x7f800000u) return (unsigned short)(u >> 16); // inf/nan passthrough
  unsigned int r = u + 0x7fffu + ((u >> 16) & 1u);
  return (unsigned short)(r >> 16);
}
static __device__ inline float bf2f(unsigned short h) {
  unsigned int u = ((unsigned int)h) << 16;
  return __builtin_bit_cast(float, u);
}
static __device__ inline v16bf cat16(v8bf lo, v8bf hi) {
  return __builtin_shufflevector(lo, hi, 0,1,2,3,4,5,6,7,8,9,10,11,12,13,14,15);
}

// A fragment 16x32 bf16 from row-major src (ISA 7.12.2 16-bit A layout):
// lane<16: row=lane,  K = {k0..k0+7, k0+16..k0+23}
// lane>=16: row=lane-16, K = {k0+8..k0+15, k0+24..k0+31}
static __device__ inline v16bf load_a16x32(const unsigned short* base, int ld,
                                           int row0, int k0, int lane) {
  int lh = lane >> 4, m = lane & 15;
  const unsigned short* p = base + (size_t)(row0 + m) * (size_t)ld + (size_t)(k0 + lh * 8);
  v8bf lo = *reinterpret_cast<const v8bf*>(p);
  v8bf hi = *reinterpret_cast<const v8bf*>(p + 16);
  return cat16(lo, hi);
}

// B fragment 32x16 bf16 loaded from B^T (row-major, row = output column n).
// lane<16: K = {k0..k0+7, k0+16..k0+23}; lane>=16: K = {k0+8..k0+15, k0+24..k0+31}.
// Each lane reads two 8-element bf16 chunks (16 bytes each) of B^T row n.
static __device__ inline v16bf load_bT32x16(const unsigned short* baseT, int ld,
                                            int n0, int k0, int lane) {
  int lh = lane >> 4, nn = lane & 15;
  const unsigned short* p = baseT + (size_t)(n0 + nn) * (size_t)ld + (size_t)(k0 + lh * 8);
  v8bf lo = *reinterpret_cast<const v8bf*>(p);
  v8bf hi = *reinterpret_cast<const v8bf*>(p + 16);
  return cat16(lo, hi);
}

static __device__ inline v8f wmma_bf(v16bf a, v16bf b, v8f c) {
  // D = A(16x32 bf16) * B(32x16 bf16) + C(16x16 f32)
  return __builtin_amdgcn_wmma_f32_16x16x32_bf16(false, a, false, b, (short)0, c, false, false);
}

static __device__ inline float rmax16(float v) {
  v = fmaxf(v, __shfl_xor(v, 1, 32));
  v = fmaxf(v, __shfl_xor(v, 2, 32));
  v = fmaxf(v, __shfl_xor(v, 4, 32));
  v = fmaxf(v, __shfl_xor(v, 8, 32));
  return v;
}
static __device__ inline float rsum16(float v) {
  v += __shfl_xor(v, 1, 32);
  v += __shfl_xor(v, 2, 32);
  v += __shfl_xor(v, 4, 32);
  v += __shfl_xor(v, 8, 32);
  return v;
}

// ---------------- Kernel 1: precision convert + weight transpose ----------------
__global__ __launch_bounds__(256) void convert_kernel(
    const float* __restrict__ hidden, const float* __restrict__ Wq,
    const float* __restrict__ Wk, const float* __restrict__ Wv,
    const float* __restrict__ Wo, unsigned short* __restrict__ hs,
    unsigned short* __restrict__ Wt, unsigned short* __restrict__ Wot) {
  size_t i = (size_t)blockIdx.x * blockDim.x + threadIdx.x;
  const size_t N1 = (size_t)S_LEN * E_DIM;
  const size_t N2 = (size_t)NALL * E_DIM;
  const size_t N3 = (size_t)E_DIM * E_DIM;
  if (i < N1) { hs[i] = f2bf(hidden[i]); return; }
  i -= N1;
  if (i < N2) {
    size_t n = i / E_DIM, e = i % E_DIM;
    float v;
    if (n < 3072)      v = Wq[e * 3072 + n];
    else if (n < 4096) v = Wk[e * 1024 + (n - 3072)];
    else               v = Wv[e * 1024 + (n - 4096)];
    Wt[i] = f2bf(v);
    return;
  }
  i -= N2;
  if (i < N3) {
    size_t n = i / E_DIM, kx = i % E_DIM;
    Wot[i] = f2bf(Wo[kx * E_DIM + n]);
  }
}

// ---------------- Kernel 2: fused QKV GEMM (64x64 tile/wave, bf16 WMMA) ----------------
// C = hs(4096x3072) @ Wqkv(3072x5120). Per K-step: 4 A-frags + 4 B-frags -> 16 WMMA
// (1:1 b128:wmma) + 8 global_prefetch of the panel 4 K-steps ahead.
__global__ __launch_bounds__(128) void qkv_gemm_kernel(
    const unsigned short* __restrict__ hs, const unsigned short* __restrict__ Wt,
    unsigned short* __restrict__ qb, unsigned short* __restrict__ kb,
    unsigned short* __restrict__ vt) {
  int lane = threadIdx.x & 31;
  int lh = lane >> 4, nn = lane & 15;
  int wid = ((int)blockIdx.x * (int)blockDim.x + (int)threadIdx.x) >> 5;
  int rt = wid / (NALL / 64);          // 64-row block (0..63)
  int cs = wid % (NALL / 64);          // 64-col strip (0..79)
  int r0 = rt * 64, n0 = cs * 64;
  v8f vz = {0.f,0.f,0.f,0.f,0.f,0.f,0.f,0.f};
  v8f acc[4][4];
#pragma unroll
  for (int rr = 0; rr < 4; rr++)
#pragma unroll
    for (int t = 0; t < 4; t++) acc[rr][t] = vz;

  for (int e0 = 0; e0 < E_DIM; e0 += 32) {
    // prefetch panels 4 K-steps ahead (lanes fan out over the 64 rows of each panel)
#pragma unroll
    for (int rr = 0; rr < 4; rr++)
      __builtin_prefetch(hs + (size_t)(r0 + rr * 16 + nn) * E_DIM + e0 + 128 + lh * 8, 0, 3);
#pragma unroll
    for (int t = 0; t < 4; t++)
      __builtin_prefetch(Wt + (size_t)(n0 + t * 16 + nn) * E_DIM + e0 + 128 + lh * 16, 0, 3);

    v16bf a[4];
#pragma unroll
    for (int rr = 0; rr < 4; rr++)
      a[rr] = load_a16x32(hs, E_DIM, r0 + rr * 16, e0, lane);
#pragma unroll
    for (int t = 0; t < 4; t++) {
      v16bf b = load_bT32x16(Wt, E_DIM, n0 + t * 16, e0, lane);
#pragma unroll
      for (int rr = 0; rr < 4; rr++)
        acc[rr][t] = wmma_bf(a[rr], b, acc[rr][t]);
    }
  }
#pragma unroll
  for (int rr = 0; rr < 4; rr++) {
#pragma unroll
    for (int t = 0; t < 4; t++) {
#pragma unroll
      for (int r = 0; r < 8; r++) {
        int row = r0 + rr * 16 + r + lh * 8;
        int col = n0 + t * 16 + nn;
        unsigned short v = f2bf(acc[rr][t][r]);
        if (col < 3072)      qb[(size_t)row * 3072 + col] = v;
        else if (col < NQK)  kb[(size_t)row * 1024 + (col - 3072)] = v;
        else                 vt[(size_t)(col - 4096) * S_LEN + row] = v;  // transposed
      }
    }
  }
}

// ---------------- Kernel 3: in-place RoPE on q and k ----------------
__global__ __launch_bounds__(256) void rope_kernel(
    const int* __restrict__ pids, unsigned short* __restrict__ qb,
    unsigned short* __restrict__ kb) {
  int i = (int)blockIdx.x * (int)blockDim.x + (int)threadIdx.x;  // S * 16 heads * 128
  int d = i & 127;
  int rest = i >> 7;
  int hh = rest & 15;
  int s = rest >> 4;
  float pos = (float)pids[s];
  float ang = pos * __expf(-(float)d * (1.0f / 128.0f) * __logf(THETA_F));
  float c = __cosf(ang), sn = __sinf(ang);
  unsigned short* buf;
  int ld, col;
  if (hh < HQ) { buf = qb; ld = 3072; col = hh * DHEAD + d; }
  else         { buf = kb; ld = 1024; col = (hh - HQ) * DHEAD + d; }
  size_t idx = (size_t)s * ld + col;
  float x0 = bf2f(buf[idx]);
  float x1 = bf2f(buf[idx + 128]);
  buf[idx]       = f2bf(x0 * c - x1 * sn);
  buf[idx + 128] = f2bf(x1 * c + x0 * sn);
}

// ---------------- Kernel 4: flash attention (causal, GQA) ----------------
// One wave per (16-query tile, head). 32 keys per step: 16 QK-WMMA + 16 PV-WMMA,
// plus prefetch of the next step's K/V tiles.
__global__ __launch_bounds__(32) void attn_kernel(
    const unsigned short* __restrict__ qb, const unsigned short* __restrict__ kb,
    const unsigned short* __restrict__ vt, const int* __restrict__ amask,
    unsigned short* __restrict__ attn) {
  const int lane = threadIdx.x & 31;
  const int qt = blockIdx.x;
  const int h = blockIdx.y;
  const int h2 = h / (HQ / HKV);
  const int q0 = qt * 16;
  const int lh = lane >> 4, nn = lane & 15;
  const float scale = 0.0625f;  // 1/sqrt(256)
  const float NEG = -1.0e30f;

  __shared__ alignas(16) unsigned short pSh[16 * 32];

  v16bf qf[8];
#pragma unroll
  for (int kk = 0; kk < 8; kk++)
    qf[kk] = load_a16x32(qb, E_DIM, q0, h * DHEAD + kk * 32, lane);

  v8f vz = {0.f,0.f,0.f,0.f,0.f,0.f,0.f,0.f};
  v8f o[16];
#pragma unroll
  for (int t = 0; t < 16; t++) o[t] = vz;
  float mi[8], li[8];
#pragma unroll
  for (int r = 0; r < 8; r++) { mi[r] = NEG; li[r] = 0.f; }

  const int steps = (q0 + 16 + 31) >> 5;
  for (int j = 0; j < steps; j++) {
    const int kbse = j * 32;

    // prefetch next step's K tile (32 rows x 512B; 32 lanes x 4 lines)
    {
      const unsigned short* pk =
          kb + (size_t)(kbse + 32 + nn + lh * 16) * 1024 + h2 * DHEAD;
#pragma unroll
      for (int q = 0; q < 4; q++) __builtin_prefetch(pk + q * 64, 0, 3);
      // prefetch next step's V tile (256 rows x 64B; 32 lanes x 8 rows)
      const unsigned short* pv =
          vt + (size_t)(h2 * DHEAD + lane * 8) * S_LEN + kbse + 32;
#pragma unroll
      for (int q = 0; q < 8; q++) __builtin_prefetch(pv + (size_t)q * S_LEN, 0, 3);
    }

    v8f s0 = vz, s1 = vz;
#pragma unroll
    for (int kk = 0; kk < 8; kk++) {
      v16bf b0 = load_bT32x16(kb, 1024, kbse,      h2 * DHEAD + kk * 32, lane);
      v16bf b1 = load_bT32x16(kb, 1024, kbse + 16, h2 * DHEAD + kk * 32, lane);
      s0 = wmma_bf(qf[kk], b0, s0);
      s1 = wmma_bf(qf[kk], b1, s1);
    }
    const int c0 = kbse + nn, c1 = c0 + 16;
    const float bias0 = (amask[c0] > 0) ? 0.f : NEG;
    const float bias1 = (amask[c1] > 0) ? 0.f : NEG;
    float alpha[8];
#pragma unroll
    for (int r = 0; r < 8; r++) {
      const int row = q0 + r + lh * 8;
      float v0 = s0[r] * scale + ((c0 <= row) ? bias0 : NEG);
      float v1 = s1[r] * scale + ((c1 <= row) ? bias1 : NEG);
      float mx = rmax16(fmaxf(v0, v1));
      float mnew = fmaxf(mi[r], mx);
      float p0 = __expf(v0 - mnew);
      float p1 = __expf(v1 - mnew);
      float sum = rsum16(p0 + p1);
      alpha[r] = __expf(mi[r] - mnew);
      li[r] = li[r] * alpha[r] + sum;
      mi[r] = mnew;
      pSh[(r + lh * 8) * 32 + nn]      = f2bf(p0);
      pSh[(r + lh * 8) * 32 + 16 + nn] = f2bf(p1);
    }
    __syncthreads();
    v16bf pf;
    {
      const unsigned short* pp = pSh + nn * 32 + lh * 8;  // A-layout read of P
      v8bf lo = *reinterpret_cast<const v8bf*>(pp);
      v8bf hi = *reinterpret_cast<const v8bf*>(pp + 16);
      pf = cat16(lo, hi);
    }
    __syncthreads();
#pragma unroll
    for (int t = 0; t < 16; t++) {
      v8f tmp = o[t];
#pragma unroll
      for (int r = 0; r < 8; r++) tmp[r] *= alpha[r];
      v16bf vb = load_bT32x16(vt, S_LEN, h2 * DHEAD + t * 16, kbse, lane);
      o[t] = wmma_bf(pf, vb, tmp);
    }
  }
#pragma unroll
  for (int t = 0; t < 16; t++) {
#pragma unroll
    for (int r = 0; r < 8; r++) {
      int row = q0 + r + lh * 8;
      int col = h * DHEAD + t * 16 + nn;
      attn[(size_t)row * E_DIM + col] = f2bf(o[t][r] / li[r]);
    }
  }
}

// ---------------- Kernel 5: output projection (64x64 tile/wave, bf16 WMMA -> f32) ----------------
__global__ __launch_bounds__(128) void out_gemm_kernel(
    const unsigned short* __restrict__ attn, const unsigned short* __restrict__ Wot,
    float* __restrict__ out) {
  int lane = threadIdx.x & 31;
  int lh = lane >> 4, nn = lane & 15;
  int wid = ((int)blockIdx.x * (int)blockDim.x + (int)threadIdx.x) >> 5;
  int rt = wid / (E_DIM / 64);
  int cs = wid % (E_DIM / 64);
  int r0 = rt * 64, n0 = cs * 64;
  v8f vz = {0.f,0.f,0.f,0.f,0.f,0.f,0.f,0.f};
  v8f acc[4][4];
#pragma unroll
  for (int rr = 0; rr < 4; rr++)
#pragma unroll
    for (int t = 0; t < 4; t++) acc[rr][t] = vz;

  for (int e0 = 0; e0 < E_DIM; e0 += 32) {
#pragma unroll
    for (int rr = 0; rr < 4; rr++)
      __builtin_prefetch(attn + (size_t)(r0 + rr * 16 + nn) * E_DIM + e0 + 128 + lh * 8, 0, 3);
#pragma unroll
    for (int t = 0; t < 4; t++)
      __builtin_prefetch(Wot + (size_t)(n0 + t * 16 + nn) * E_DIM + e0 + 128 + lh * 16, 0, 3);

    v16bf a[4];
#pragma unroll
    for (int rr = 0; rr < 4; rr++)
      a[rr] = load_a16x32(attn, E_DIM, r0 + rr * 16, e0, lane);
#pragma unroll
    for (int t = 0; t < 4; t++) {
      v16bf b = load_bT32x16(Wot, E_DIM, n0 + t * 16, e0, lane);
#pragma unroll
      for (int rr = 0; rr < 4; rr++)
        acc[rr][t] = wmma_bf(a[rr], b, acc[rr][t]);
    }
  }
#pragma unroll
  for (int rr = 0; rr < 4; rr++) {
#pragma unroll
    for (int t = 0; t < 4; t++) {
#pragma unroll
      for (int r = 0; r < 8; r++) {
        int row = r0 + rr * 16 + r + lh * 8;
        int col = n0 + t * 16 + nn;
        out[(size_t)row * E_DIM + col] = acc[rr][t][r];
      }
    }
  }
}

// ---------------- workspace layout (bytes) ----------------
static constexpr size_t OFF_HS  = 0;
static constexpr size_t SZ_HS   = (size_t)S_LEN * E_DIM * 2;   // 25.2 MB
static constexpr size_t OFF_WT  = OFF_HS + SZ_HS;
static constexpr size_t SZ_WT   = (size_t)NALL * E_DIM * 2;    // 31.5 MB
static constexpr size_t OFF_WOT = OFF_WT + SZ_WT;
static constexpr size_t SZ_WOT  = (size_t)E_DIM * E_DIM * 2;   // 18.9 MB
static constexpr size_t OFF_Q   = OFF_WOT + SZ_WOT;
static constexpr size_t SZ_Q    = (size_t)S_LEN * E_DIM * 2;   // 25.2 MB
static constexpr size_t OFF_K   = OFF_Q + SZ_Q;
static constexpr size_t SZ_K    = (size_t)S_LEN * 1024 * 2;    // 8.4 MB
static constexpr size_t OFF_VT  = OFF_K + SZ_K;
static constexpr size_t SZ_VT   = SZ_K;                        // 8.4 MB
static constexpr size_t OFF_AT  = OFF_VT + SZ_VT;
// total ~142.6 MB

extern "C" void kernel_launch(void* const* d_in, const int* in_sizes, int n_in,
                              void* d_out, int out_size, void* d_ws, size_t ws_size,
                              hipStream_t stream) {
  const float* hidden = (const float*)d_in[0];
  const int*   amask  = (const int*)d_in[1];
  const int*   pids   = (const int*)d_in[2];
  const float* Wq     = (const float*)d_in[3];
  const float* Wk     = (const float*)d_in[4];
  const float* Wv     = (const float*)d_in[5];
  const float* Wo     = (const float*)d_in[6];
  float* out = (float*)d_out;

  char* ws = (char*)d_ws;
  unsigned short* hs  = (unsigned short*)(ws + OFF_HS);
  unsigned short* Wt  = (unsigned short*)(ws + OFF_WT);
  unsigned short* Wot = (unsigned short*)(ws + OFF_WOT);
  unsigned short* qb  = (unsigned short*)(ws + OFF_Q);
  unsigned short* kb  = (unsigned short*)(ws + OFF_K);
  unsigned short* vt  = (unsigned short*)(ws + OFF_VT);
  unsigned short* at  = (unsigned short*)(ws + OFF_AT);

  // 1) convert + transpose weights: 37,748,736 elements
  {
    size_t total = (size_t)S_LEN * E_DIM + (size_t)NALL * E_DIM + (size_t)E_DIM * E_DIM;
    int blocks = (int)((total + 255) / 256);
    convert_kernel<<<blocks, 256, 0, stream>>>(hidden, Wq, Wk, Wv, Wo, hs, Wt, Wot);
  }
  // 2) QKV GEMM: 64 row blocks * 80 col strips = 5120 waves, 4 waves/block
  qkv_gemm_kernel<<<(64 * (NALL / 64)) / 4, 128, 0, stream>>>(hs, Wt, qb, kb, vt);
  // 3) RoPE: 4096 * 16 heads * 128 pairs = 8,388,608 threads
  rope_kernel<<<(S_LEN * 16 * 128) / 256, 256, 0, stream>>>(pids, qb, kb);
  // 4) flash attention: (256 query tiles, 12 heads), 1 wave per block
  attn_kernel<<<dim3(S_LEN / 16, HQ), 32, 0, stream>>>(qb, kb, vt, amask, at);
  // 5) output projection: 64 * 48 = 3072 waves, 4 waves/block
  out_gemm_kernel<<<(64 * (E_DIM / 64)) / 4, 128, 0, stream>>>(at, Wot, out);
}
